// GQA_22694607192162
// MI455X (gfx1250) — matrix operand order repaired and checked by reading, not compiled, not run on hardware
//
#include <hip/hip_runtime.h>
#include <cmath>

typedef __attribute__((ext_vector_type(16))) __bf16 v16bf;
typedef __attribute__((ext_vector_type(8)))  float  v8f;
typedef __attribute__((ext_vector_type(4)))  unsigned int u32x4;
typedef __attribute__((ext_vector_type(8)))  int i32x8;
typedef __attribute__((ext_vector_type(4)))  int i32x4;

#if defined(__AMDGCN__) && __has_builtin(__builtin_amdgcn_tensor_load_to_lds) && \
    __has_builtin(__builtin_amdgcn_s_wait_tensorcnt)
#define USE_TDM 1
#else
#define USE_TDM 0
#endif

namespace {

constexpr int Dm  = 1024;            // d_model
constexpr int Gg  = 8;               // KV groups
constexpr int Hh  = 16;              // heads per group
constexpr int DQk = 128;             // qk head dim
constexpr int DVv = 128;             // v head dim
constexpr int FFf = 1024;            // ff out
constexpr int Ss  = 1024;            // sequence
constexpr int NQ  = Gg * Hh * DQk;   // 16384
constexpr int NKV = Gg * (DQk + DVv);// 2048
constexpr int DZ  = Gg * Hh * DVv;   // 16384

struct alignas(16) B16 { unsigned int w[4]; };   // POD 16-byte chunk
union Frag { v16bf v; B16 q[2]; };               // WMMA 16-bit operand (8 VGPRs)

__device__ inline __bf16 f2bf(float f) {
  union { float f; unsigned u; } a; a.f = f;
  unsigned r = (a.u + 0x7FFFu + ((a.u >> 16) & 1u)) >> 16;   // RNE
  union { unsigned short s; __bf16 h; } b; b.s = (unsigned short)r;
  return b.h;
}

__device__ inline float silu_f(float v) { return v / (1.f + __expf(-v)); }

__device__ inline v8f wmma_bf16(const Frag& a, const Frag& b, v8f c) {
  return __builtin_amdgcn_wmma_f32_16x16x32_bf16(
      false, a.v, false, b.v, (short)0, c, false, false);
}

#if USE_TDM
// Byte offset of a generic pointer inside the workgroup LDS allocation.
__device__ inline unsigned lds_off_of(const void* p) {
  return (unsigned)(__attribute__((address_space(3))) const char*)p;
}

// Issue one TDM 2D tile load: tile_w x tile_h elements (bf16), global row
// stride gstride elements, LDS destination padded by (pamt+1) DWORDs every
// 2^pint * 8 bytes (hardware pad -> bank-conflict-free LDS strides).
__device__ inline void tdm_load_tile(const void* gsrc, const void* ldst,
                                     unsigned tw, unsigned th,
                                     unsigned gstride, unsigned pint, unsigned pamt)
{
  unsigned long long ga = (unsigned long long)(size_t)gsrc;
  u32x4 g0;
  g0[0] = 1u;                                            // count=1 (valid), user mode
  g0[1] = lds_off_of(ldst);                              // lds_addr (bytes)
  g0[2] = (unsigned)ga;                                  // global_addr[31:0]
  g0[3] = ((unsigned)(ga >> 32) & 0x01FFFFFFu) | (2u << 30);  // addr[56:32] | type=2
  i32x8 g1;
  g1[0] = (int)((1u << 16)                               // data_size = 2 bytes
              | (1u << 20)                               // pad_enable
              | (pint << 22) | (pamt << 25));            // pad interval / amount
  g1[1] = (int)((gstride & 0xFFFFu) << 16);              // tensor_dim0[15:0]
  g1[2] = (int)((gstride >> 16) & 0xFFFFu);              // tensor_dim0[31:16], dim1 lo=0
  g1[3] = (int)(16u | (tw << 16));                       // tensor_dim1=1<<20 ; tile_dim0
  g1[4] = (int)(th & 0xFFFFu);                           // tile_dim1 ; tile_dim2=0
  g1[5] = (int)gstride;                                  // tensor_dim0_stride[31:0]
  g1[6] = 0;                                             // stride hi / dim1_stride lo
  g1[7] = 0;
  i32x4 z4 = {0, 0, 0, 0};
  i32x8 z8 = {0, 0, 0, 0, 0, 0, 0, 0};
  // clang-23 / therock-10.0 form: (g0, g1, g2, g3, g4?, cpol) -- 6 args
  __builtin_amdgcn_tensor_load_to_lds(g0, g1, z4, z4, z8, 0);
}
#endif

// -------------------------------------------------------------------------
// fp32 -> bf16 cast (8 elements per thread)
// -------------------------------------------------------------------------
__global__ __launch_bounds__(256) void cast_bf16_kernel(
    const float* __restrict__ in, __bf16* __restrict__ out, int n8)
{
  int i = blockIdx.x * 256 + threadIdx.x;
  if (i >= n8) return;
  const float* p = in + (size_t)i * 8;
  union { __bf16 h[8]; B16 q; } u;
#pragma unroll
  for (int j = 0; j < 8; ++j) u.h[j] = f2bf(p[j]);
  *(B16*)(out + (size_t)i * 8) = u.q;
}

// -------------------------------------------------------------------------
// RMSNorm: out_bf16[row,:] = x / max(||x||,1e-12) * sqrt(cols) * gamma
// -------------------------------------------------------------------------
__global__ __launch_bounds__(256) void rmsnorm_bf16_kernel(
    const float* __restrict__ x, const float* __restrict__ gamma,
    __bf16* __restrict__ out, int cols, float sqrtd)
{
  const int row = blockIdx.x;
  const float* xr = x + (size_t)row * cols;
  float ss = 0.f;
  for (int c = threadIdx.x; c < cols; c += 256) { float v = xr[c]; ss += v * v; }
  __shared__ float red[8];
#pragma unroll
  for (int d = 16; d; d >>= 1) ss += __shfl_xor(ss, d, 32);
  if ((threadIdx.x & 31) == 0) red[threadIdx.x >> 5] = ss;
  __syncthreads();
  float tot = 0.f;
#pragma unroll
  for (int w = 0; w < 8; ++w) tot += red[w];
  const float inv = sqrtd / fmaxf(sqrtf(tot), 1e-12f);
  __bf16* orow = out + (size_t)row * cols;
  for (int c = threadIdx.x; c < cols; c += 256)
    orow[c] = f2bf(xr[c] * inv * gamma[c]);
}

// -------------------------------------------------------------------------
// NT GEMM: C(M,N) = A(M,K) * B(N,K)^T + bias, bf16 in, fp32 accumulate.
// 128x128 block tile, BK=64, double-buffered LDS, TDM staging when available.
// 8 waves in 4x2 grid, each wave 32x64 output (2x4 WMMA tiles).
// MODE: 0 silu->fp32 (FFN2) | 1 silu->bf16 (FFN1)
//       2 Q scatter (head,S,128) bf16 | 3 KV split: K (g,S,128), V^T (g,128,S)
// -------------------------------------------------------------------------
template<int MODE>
__global__ __launch_bounds__(256) void gemm_nt_kernel(
    const __bf16* __restrict__ A, const __bf16* __restrict__ Bw,
    const float* __restrict__ bias, void* __restrict__ out0,
    void* __restrict__ out1, int N, int K)
{
  constexpr int LDT = 72;                 // 64 + 8 pad -> conflict-free b128
  __shared__ __bf16 As[2][128 * LDT];
  __shared__ __bf16 Bs[2][128 * LDT];
  const int tid  = threadIdx.x;
  const int wave = tid >> 5, lane = tid & 31;
  const int r = lane & 15, hf = lane >> 4;
  const int wm = wave & 3, wn = wave >> 2;
  const int bm = blockIdx.y * 128, bn = blockIdx.x * 128;

  auto stage = [&](int buf, int k0) {
#if USE_TDM
    if (tid < 32) {                       // one TDM descriptor pair per tile
      tdm_load_tile(&A [(size_t)bm * K + k0], &As[buf][0], 64, 128, (unsigned)K, 4, 3);
      tdm_load_tile(&Bw[(size_t)bn * K + k0], &Bs[buf][0], 64, 128, (unsigned)K, 4, 3);
    }
#else
    for (int c = tid; c < 1024; c += 256) {
      int row = c >> 3, col = (c & 7) * 8;
      *(B16*)&As[buf][row * LDT + col] = *(const B16*)&A [(size_t)(bm + row) * K + k0 + col];
      *(B16*)&Bs[buf][row * LDT + col] = *(const B16*)&Bw[(size_t)(bn + row) * K + k0 + col];
    }
#endif
  };

  v8f acc[2][4] = {};
  const int nk = K >> 6;
  stage(0, 0);
  for (int i = 0; i < nk; ++i) {
    const int p = i & 1;
#if USE_TDM
    __builtin_amdgcn_s_wait_tensorcnt((short)0);
#endif
    __syncthreads();                      // buffer p ready; readers of 1-p done
    if (i + 1 < nk) stage(1 - p, (i + 1) * 64);
    const __bf16* Asp = &As[p][0];
    const __bf16* Bsp = &Bs[p][0];
#pragma unroll
    for (int kc = 0; kc < 2; ++kc) {
      const int ko = kc * 32;
      Frag a0, a1, b0, b1, b2, b3;        // batch all frag loads, then WMMA burst
      { const __bf16* q = Asp + (wm * 32 + r) * LDT + ko;
        a0.q[0] = *(const B16*)(q + hf * 16); a0.q[1] = *(const B16*)(q + hf * 16 + 8); }
      { const __bf16* q = Asp + (wm * 32 + 16 + r) * LDT + ko;
        a1.q[0] = *(const B16*)(q + hf * 16); a1.q[1] = *(const B16*)(q + hf * 16 + 8); }
      { const __bf16* q = Bsp + (wn * 64 +  0 + r) * LDT + ko + hf * 16;
        b0.q[0] = *(const B16*)q; b0.q[1] = *(const B16*)(q + 8); }
      { const __bf16* q = Bsp + (wn * 64 + 16 + r) * LDT + ko + hf * 16;
        b1.q[0] = *(const B16*)q; b1.q[1] = *(const B16*)(q + 8); }
      { const __bf16* q = Bsp + (wn * 64 + 32 + r) * LDT + ko + hf * 16;
        b2.q[0] = *(const B16*)q; b2.q[1] = *(const B16*)(q + 8); }
      { const __bf16* q = Bsp + (wn * 64 + 48 + r) * LDT + ko + hf * 16;
        b3.q[0] = *(const B16*)q; b3.q[1] = *(const B16*)(q + 8); }
      acc[0][0] = wmma_bf16(a0, b0, acc[0][0]);
      acc[1][0] = wmma_bf16(a1, b0, acc[1][0]);
      acc[0][1] = wmma_bf16(a0, b1, acc[0][1]);
      acc[1][1] = wmma_bf16(a1, b1, acc[1][1]);
      acc[0][2] = wmma_bf16(a0, b2, acc[0][2]);
      acc[1][2] = wmma_bf16(a1, b2, acc[1][2]);
      acc[0][3] = wmma_bf16(a0, b3, acc[0][3]);
      acc[1][3] = wmma_bf16(a1, b3, acc[1][3]);
    }
  }

  const int ln = r;
#pragma unroll
  for (int mt = 0; mt < 2; ++mt)
#pragma unroll
    for (int nt = 0; nt < 4; ++nt)
#pragma unroll
      for (int i = 0; i < 8; ++i) {
        int gm = bm + wm * 32 + mt * 16 + (hf ? i + 8 : i);
        int gn = bn + wn * 64 + nt * 16 + ln;
        float v = acc[mt][nt][i] + bias[gn];
        if constexpr (MODE == 0) {
          ((float*)out0)[(size_t)gm * N + gn] = silu_f(v);
        } else if constexpr (MODE == 1) {
          ((__bf16*)out0)[(size_t)gm * N + gn] = f2bf(silu_f(v));
        } else if constexpr (MODE == 2) {
          int head = gn >> 7, d = gn & 127;
          ((__bf16*)out0)[((size_t)head * Ss + gm) * DQk + d] = f2bf(v);
        } else {
          if (gn < Gg * DQk) {
            int gi = gn >> 7, d = gn & 127;
            ((__bf16*)out0)[((size_t)gi * Ss + gm) * DQk + d] = f2bf(v);
          } else {
            int gn2 = gn - Gg * DQk;
            int gi = gn2 >> 7, d = gn2 & 127;
            ((__bf16*)out1)[((size_t)gi * DVv + d) * Ss + gm] = f2bf(v);
          }
        }
      }
}

// -------------------------------------------------------------------------
// Flash attention: grid (S/64, G*H), 128 threads (4 waves x 16 query rows).
// Double-buffered K / V^T tiles (TDM staged), grouped fragment preloads,
// online softmax in WMMA C-layout, P re-laid-out through per-wave LDS.
// -------------------------------------------------------------------------
__global__ __launch_bounds__(128) void flash_attn_kernel(
    const __bf16* __restrict__ Q, const __bf16* __restrict__ K,
    const __bf16* __restrict__ Vt, float* __restrict__ z)
{
  constexpr int LDK = 136;          // 128 + 8 pad
  constexpr int LDV = 40;           // 32 + 8 pad
  __shared__ __bf16 Ks[2][32 * LDK];
  __shared__ __bf16 Vs[2][128 * LDV];
  __shared__ __bf16 Ps[4 * 16 * LDV];
  const int tid  = threadIdx.x;
  const int wave = tid >> 5, lane = tid & 31;
  const int r = lane & 15, hf = lane >> 4, ln = r;
  const int hg = blockIdx.y;
  const int g  = hg >> 4;
  const int m0 = blockIdx.x * 64 + wave * 16;
  const __bf16* Qh = Q  + (size_t)hg * Ss * DQk;
  const __bf16* Kg = K  + (size_t)g  * Ss * DQk;
  const __bf16* Vg = Vt + (size_t)g  * DVv * Ss;

  Frag qf[4];                                    // resident 16 x 128 Q tile
#pragma unroll
  for (int kk = 0; kk < 4; ++kk) {
    const __bf16* p = Qh + (size_t)(m0 + r) * DQk + kk * 32;
    qf[kk].q[0] = *(const B16*)(p + hf * 16);
    qf[kk].q[1] = *(const B16*)(p + hf * 16 + 8);
  }

  auto stage = [&](int buf, int t0) {
#if USE_TDM
    if (tid < 32) {
      tdm_load_tile(&Kg[(size_t)t0 * DQk], &Ks[buf][0], 128, 32, DQk, 5, 3);
      tdm_load_tile(&Vg[t0],               &Vs[buf][0], 32, 128, Ss, 3, 3);
    }
#else
    for (int c = tid; c < 512; c += 128) {
      int row = c >> 4, col = (c & 15) * 8;
      *(B16*)&Ks[buf][row * LDK + col] = *(const B16*)&Kg[(size_t)(t0 + row) * DQk + col];
    }
    for (int c = tid; c < 512; c += 128) {
      int row = c >> 2, col = (c & 3) * 8;
      *(B16*)&Vs[buf][row * LDV + col] = *(const B16*)&Vg[(size_t)row * Ss + t0 + col];
    }
#endif
  };

  v8f O[8] = {};
  float mrun[8], lrun[8];
#pragma unroll
  for (int i = 0; i < 8; ++i) { mrun[i] = -3.0e38f; lrun[i] = 0.f; }
  const float scale = 0.088388347648318447f;     // 1/sqrt(128)
  __bf16* Pw = Ps + wave * (16 * LDV);

  stage(0, 0);
  for (int it = 0; it < Ss / 32; ++it) {
    const int t0 = it * 32;
    const int p  = it & 1;
#if USE_TDM
    __builtin_amdgcn_s_wait_tensorcnt((short)0);
#endif
    __syncthreads();
    if (t0 + 32 < Ss) stage(1 - p, t0 + 32);
    const __bf16* Kp = &Ks[p][0];
    const __bf16* Vp = &Vs[p][0];

    // scores: 16 x 32 = Q(16x128) * K_blk(32x128)^T, frag loads batched
    v8f s0 = {}, s1 = {};
#pragma unroll
    for (int g2 = 0; g2 < 2; ++g2) {
      const int kc = g2 * 64;
      Frag b00, b01, b10, b11;
      { const __bf16* q = Kp + (0  + r) * LDK + kc      + hf * 16;
        b00.q[0] = *(const B16*)q; b00.q[1] = *(const B16*)(q + 8); }
      { const __bf16* q = Kp + (16 + r) * LDK + kc      + hf * 16;
        b01.q[0] = *(const B16*)q; b01.q[1] = *(const B16*)(q + 8); }
      { const __bf16* q = Kp + (0  + r) * LDK + kc + 32 + hf * 16;
        b10.q[0] = *(const B16*)q; b10.q[1] = *(const B16*)(q + 8); }
      { const __bf16* q = Kp + (16 + r) * LDK + kc + 32 + hf * 16;
        b11.q[0] = *(const B16*)q; b11.q[1] = *(const B16*)(q + 8); }
      s0 = wmma_bf16(qf[2 * g2],     b00, s0);
      s1 = wmma_bf16(qf[2 * g2],     b01, s1);
      s0 = wmma_bf16(qf[2 * g2 + 1], b10, s0);
      s1 = wmma_bf16(qf[2 * g2 + 1], b11, s1);
    }

    // online softmax (row = acc index i within the half-wave)
#pragma unroll
    for (int i = 0; i < 8; ++i) {
      float a0 = s0[i] * scale, a1 = s1[i] * scale;
      float mx = fmaxf(a0, a1);
#pragma unroll
      for (int d = 1; d < 16; d <<= 1) mx = fmaxf(mx, __shfl_xor(mx, d, 32));
      float mnew  = fmaxf(mrun[i], mx);
      float alpha = __expf(mrun[i] - mnew);
      float p0 = __expf(a0 - mnew), p1 = __expf(a1 - mnew);
      float rs = p0 + p1;
#pragma unroll
      for (int d = 1; d < 16; d <<= 1) rs += __shfl_xor(rs, d, 32);
      lrun[i] = lrun[i] * alpha + rs;
      mrun[i] = mnew;
#pragma unroll
      for (int j = 0; j < 8; ++j) O[j][i] = O[j][i] * alpha;
      int rowm = hf ? (i + 8) : i;
      Pw[rowm * LDV + ln]      = f2bf(p0);
      Pw[rowm * LDV + 16 + ln] = f2bf(p1);
    }

    // P (C-layout) -> A-layout via per-wave LDS (same-wave DS ops in order)
    Frag pf;
    { const __bf16* q = Pw + r * LDV;
      pf.q[0] = *(const B16*)(q + hf * 16);
      pf.q[1] = *(const B16*)(q + hf * 16 + 8); }

    // O += P(16x32) * V_blk(32x128) : 2 groups of 4 batched V-frag loads
#pragma unroll
    for (int g2 = 0; g2 < 2; ++g2) {
      Frag v0, v1, v2, v3;
      const int j0 = g2 * 4;
      { const __bf16* q = Vp + ((j0 + 0) * 16 + r) * LDV + hf * 16;
        v0.q[0] = *(const B16*)q; v0.q[1] = *(const B16*)(q + 8); }
      { const __bf16* q = Vp + ((j0 + 1) * 16 + r) * LDV + hf * 16;
        v1.q[0] = *(const B16*)q; v1.q[1] = *(const B16*)(q + 8); }
      { const __bf16* q = Vp + ((j0 + 2) * 16 + r) * LDV + hf * 16;
        v2.q[0] = *(const B16*)q; v2.q[1] = *(const B16*)(q + 8); }
      { const __bf16* q = Vp + ((j0 + 3) * 16 + r) * LDV + hf * 16;
        v3.q[0] = *(const B16*)q; v3.q[1] = *(const B16*)(q + 8); }
      O[j0 + 0] = wmma_bf16(pf, v0, O[j0 + 0]);
      O[j0 + 1] = wmma_bf16(pf, v1, O[j0 + 1]);
      O[j0 + 2] = wmma_bf16(pf, v2, O[j0 + 2]);
      O[j0 + 3] = wmma_bf16(pf, v3, O[j0 + 3]);
    }
  }

#pragma unroll
  for (int j = 0; j < 8; ++j)
#pragma unroll
    for (int i = 0; i < 8; ++i) {
      int rowm = hf ? (i + 8) : i;
      int srow = blockIdx.x * 64 + wave * 16 + rowm;
      int col  = hg * DVv + j * 16 + ln;
      z[(size_t)srow * DZ + col] = O[j][i] / lrun[i];
    }
}

} // anonymous namespace

// -------------------------------------------------------------------------
// Host-side launch sequence
// -------------------------------------------------------------------------
extern "C" void kernel_launch(void* const* d_in, const int* in_sizes, int n_in,
                              void* d_out, int out_size, void* d_ws, size_t ws_size,
                              hipStream_t stream)
{
  (void)in_sizes; (void)n_in; (void)out_size; (void)ws_size;
  const float* x     = (const float*)d_in[0];
  const float* g_in  = (const float*)d_in[1];
  const float* Wq_w  = (const float*)d_in[2];
  const float* Wq_b  = (const float*)d_in[3];
  const float* Wkv_w = (const float*)d_in[4];
  const float* Wkv_b = (const float*)d_in[5];
  const float* g_z   = (const float*)d_in[6];
  const float* W1    = (const float*)d_in[7];
  const float* b1    = (const float*)d_in[8];
  const float* W2    = (const float*)d_in[9];
  const float* b2    = (const float*)d_in[10];
  float* out = (float*)d_out;

  char* ws = (char*)d_ws;
  size_t off = 0;
  auto take = [&](size_t bytes) -> char* {
    char* p = ws + off; off += (bytes + 255) & ~(size_t)255; return p;
  };
  __bf16* xn   = (__bf16*)take((size_t)Ss * Dm * 2);
  __bf16* WqB  = (__bf16*)take((size_t)NQ * Dm * 2);
  __bf16* WkvB = (__bf16*)take((size_t)NKV * Dm * 2);
  __bf16* W1B  = (__bf16*)take((size_t)(2 * FFf) * DZ * 2);
  __bf16* W2B  = (__bf16*)take((size_t)FFf * (2 * FFf) * 2);
  __bf16* Qb   = (__bf16*)take((size_t)Ss * NQ * 2);        // (head, S, 128)
  __bf16* Kb   = (__bf16*)take((size_t)Gg * Ss * DQk * 2);  // (g, S, 128)
  __bf16* Vtb  = (__bf16*)take((size_t)Gg * DVv * Ss * 2);  // (g, 128, S)
  float*  zbuf = (float*) take((size_t)Ss * DZ * 4);
  __bf16* zn   = (__bf16*)take((size_t)Ss * DZ * 2);
  __bf16* h1   = (__bf16*)take((size_t)Ss * (2 * FFf) * 2);

  auto castN = [&](const float* src, __bf16* dst, size_t n) {
    int n8 = (int)(n / 8);
    cast_bf16_kernel<<<n8 / 256, 256, 0, stream>>>(src, dst, n8);
  };
  castN(Wq_w,  WqB,  (size_t)NQ * Dm);
  castN(Wkv_w, WkvB, (size_t)NKV * Dm);
  castN(W1,    W1B,  (size_t)(2 * FFf) * DZ);
  castN(W2,    W2B,  (size_t)FFf * (2 * FFf));

  rmsnorm_bf16_kernel<<<Ss, 256, 0, stream>>>(x, g_in, xn, Dm, 32.0f);

  gemm_nt_kernel<2><<<dim3(NQ / 128, Ss / 128), 256, 0, stream>>>(
      xn, WqB, Wq_b, Qb, nullptr, NQ, Dm);
  gemm_nt_kernel<3><<<dim3(NKV / 128, Ss / 128), 256, 0, stream>>>(
      xn, WkvB, Wkv_b, Kb, Vtb, NKV, Dm);

  flash_attn_kernel<<<dim3(Ss / 64, Gg * Hh), 128, 0, stream>>>(Qb, Kb, Vtb, zbuf);

  rmsnorm_bf16_kernel<<<Ss, 256, 0, stream>>>(zbuf, g_z, zn, DZ, 128.0f);

  gemm_nt_kernel<1><<<dim3((2 * FFf) / 128, Ss / 128), 256, 0, stream>>>(
      zn, W1B, b1, h1, nullptr, 2 * FFf, DZ);
  gemm_nt_kernel<0><<<dim3(FFf / 128, Ss / 128), 256, 0, stream>>>(
      h1, W2B, b2, out, nullptr, FFf, 2 * FFf);
}
